// MultiHeadedAttention_3770981286599
// MI455X (gfx1250) — compile-verified, matrix operand order checked by reading, not run on hardware
//
#include <hip/hip_runtime.h>
#include <hip/hip_bf16.h>
#include <math.h>

// Problem constants (match reference)
#define B_  4
#define T_  2048
#define D_  1024
#define H_  16
#define DK_ 64

typedef __attribute__((ext_vector_type(16))) __bf16 v16bf;
typedef __attribute__((ext_vector_type(4)))  __bf16 v4bf;
typedef __attribute__((ext_vector_type(8)))  float  v8f;
typedef unsigned int v4u __attribute__((ext_vector_type(4)));
typedef int          v8i __attribute__((ext_vector_type(8)));
typedef int          v4i __attribute__((ext_vector_type(4)));

// ---------------------------------------------------------------------------
// Helpers
// ---------------------------------------------------------------------------

static __device__ __forceinline__ v8f wmma_bf16(v16bf a, v16bf b, v8f c) {
    // D = A(16x32 bf16) x B(32x16 bf16) + C(16x16 f32)
    return __builtin_amdgcn_wmma_f32_16x16x32_bf16(
        /*neg_a=*/false, a, /*neg_b=*/false, b,
        /*c_mod=*/(short)0, c, /*reuse_a=*/false, /*reuse_b=*/false);
}

// Load a 16-element bf16 fragment run (32 bytes, aligned) for one lane.
static __device__ __forceinline__ v16bf frag_bf16(const __bf16* p) {
    return *(const v16bf*)p;
}

// Load 16 consecutive floats and convert to a bf16 fragment (native cvt).
static __device__ __forceinline__ v16bf frag_f32(const float* __restrict__ p) {
    const float4* q4 = (const float4*)p;
    v16bf v;
#pragma unroll
    for (int j = 0; j < 4; ++j) {
        float4 f = q4[j];
        v[4 * j + 0] = (__bf16)f.x;
        v[4 * j + 1] = (__bf16)f.y;
        v[4 * j + 2] = (__bf16)f.z;
        v[4 * j + 3] = (__bf16)f.w;
    }
    return v;
}

// C/D 16x16 f32 tile layout (wave32): VGPR i -> M = i + 8*(lane/16), N = lane%16
#define TILE_ROW(tm, lane, i) ((tm) + (((lane) >> 4) << 3) + (i))
#define TILE_COL(tn, lane)    ((tn) + ((lane) & 15))

// ---------------------------------------------------------------------------
// Tensor Data Mover: 2-D tile DMA (global -> LDS), bf16 (2-byte) elements.
// Builds D# groups per CDNA5 ISA ch.8:
//   group0: [1:0]=count=1, [63:32]=lds_addr, [120:64]=global_addr, [127:126]=2
//   group1: data_size=1(2B), tensor_dim0/1, tile_dim0/1, tensor_dim0_stride
// ---------------------------------------------------------------------------
static __device__ __forceinline__ void tdm_load_2d_bf16(
    unsigned lds_off, const void* gptr,
    unsigned tensor_d0, unsigned tensor_d1, unsigned stride0_elems,
    unsigned tile_d0, unsigned tile_d1)
{
    unsigned long long ga = (unsigned long long)(uintptr_t)gptr;
    v4u g0;
    g0[0] = 1u;                                            // count=1 (valid)
    g0[1] = lds_off;                                       // lds_addr (bytes)
    g0[2] = (unsigned)(ga & 0xffffffffu);                  // global_addr[31:0]
    g0[3] = (unsigned)((ga >> 32) & 0x01ffffffu) | (2u << 30); // addr[56:32]|type=2

    v8i g1;
    g1[0] = (int)(1u << 16);                               // data_size=1 -> 2B
    g1[1] = (int)((tensor_d0 & 0xffffu) << 16);            // tensor_dim0[15:0]
    g1[2] = (int)(((tensor_d0 >> 16) & 0xffffu) |
                  ((tensor_d1 & 0xffffu) << 16));          // d0[31:16] | d1[15:0]
    g1[3] = (int)(((tensor_d1 >> 16) & 0xffffu) |
                  ((tile_d0 & 0xffffu) << 16));            // d1[31:16] | tile0
    g1[4] = (int)(tile_d1 & 0xffffu);                      // tile1 | tile2=0
    g1[5] = (int)stride0_elems;                            // dim0_stride[31:0]
    g1[6] = 0;                                             // stride hi bits
    g1[7] = 0;

    v4i z4 = {0, 0, 0, 0};
#if defined(__clang_major__) && (__clang_major__ >= 23)
    v8i z8 = {0, 0, 0, 0, 0, 0, 0, 0};
    __builtin_amdgcn_tensor_load_to_lds(g0, g1, z4, z4, z8, 0);
#else
    __builtin_amdgcn_tensor_load_to_lds(g0, g1, z4, z4, 0);
#endif
}

// ---------------------------------------------------------------------------
// Kernel 0: elementwise fp32 -> bf16 (vectorized by 4)
// ---------------------------------------------------------------------------
__global__ __launch_bounds__(256) void mha_cvt_kernel(
    const float* __restrict__ src, __bf16* __restrict__ dst, int n4)
{
    int i = blockIdx.x * 256 + threadIdx.x;
    if (i < n4) {
        float4 f = ((const float4*)src)[i];
        v4bf o;
        o[0] = (__bf16)f.x; o[1] = (__bf16)f.y;
        o[2] = (__bf16)f.z; o[3] = (__bf16)f.w;
        ((v4bf*)dst)[i] = o;
    }
}

// ---------------------------------------------------------------------------
// Kernel 1: fused QKV projection.  Y = X @ W^T + b  (M=B*T, N=D, K=D)
// All operands pre-converted bf16. blockIdx.z: 0->Q, 1->K, 2->V(transposed/head)
// ---------------------------------------------------------------------------
__global__ __launch_bounds__(256) void mha_qkv_kernel(
    const __bf16* __restrict__ qx, const __bf16* __restrict__ kx,
    const __bf16* __restrict__ vx,
    const __bf16* __restrict__ wqb, const __bf16* __restrict__ wkb,
    const __bf16* __restrict__ wvb,
    const float* __restrict__ bq, const float* __restrict__ bk,
    const float* __restrict__ bv,
    __bf16* __restrict__ qb, __bf16* __restrict__ kb,
    __bf16* __restrict__ vtb)
{
    const int which = blockIdx.z;
    const __bf16* X    = (which == 0) ? qx  : (which == 1) ? kx  : vx;
    const __bf16* W    = (which == 0) ? wqb : (which == 1) ? wkb : wvb;
    const float*  bias = (which == 0) ? bq  : (which == 1) ? bk  : bv;

    const int tid  = threadIdx.x;
    const int lane = tid & 31;
    const int w    = tid >> 5;              // 8 waves: 4 (M) x 2 (N)
    const int m0   = blockIdx.x * 128 + (w & 3) * 32;
    const int n0   = blockIdx.y * 64  + (w >> 2) * 32;
    const int ar   = lane & 15;
    const int kh   = (lane >> 4) * 16;

    const __bf16* Am0 = X + (size_t)(m0 + ar)      * D_;
    const __bf16* Am1 = X + (size_t)(m0 + 16 + ar) * D_;
    const __bf16* Bn0 = W + (size_t)(n0 + ar)      * D_;
    const __bf16* Bn1 = W + (size_t)(n0 + 16 + ar) * D_;

    v8f c00 = {}, c01 = {}, c10 = {}, c11 = {};
    for (int k0 = 0; k0 < D_; k0 += 32) {
        const int kk = k0 + kh;
        __builtin_prefetch(Am0 + kk + 256, 0, 0);
        __builtin_prefetch(Bn0 + kk + 256, 0, 0);
        v16bf a0 = frag_bf16(Am0 + kk);
        v16bf a1 = frag_bf16(Am1 + kk);
        v16bf b0 = frag_bf16(Bn0 + kk);
        v16bf b1 = frag_bf16(Bn1 + kk);
        c00 = wmma_bf16(a0, b0, c00);
        c01 = wmma_bf16(a0, b1, c01);
        c10 = wmma_bf16(a1, b0, c10);
        c11 = wmma_bf16(a1, b1, c11);
    }

    v8f acc[4] = { c00, c01, c10, c11 };
    int tms[4] = { m0, m0, m0 + 16, m0 + 16 };
    int tns[4] = { n0, n0 + 16, n0, n0 + 16 };

#pragma unroll
    for (int t = 0; t < 4; ++t) {
        const int col = TILE_COL(tns[t], lane);
        const float bvv = bias[col];
        if (which == 2) {
            const int h  = col >> 6;
            const int dk = col & (DK_ - 1);
#pragma unroll
            for (int i = 0; i < 8; ++i) {
                const int row = TILE_ROW(tms[t], lane, i);   // row = b*T + t
                const int bb  = row >> 11;
                const int tt  = row & (T_ - 1);
                vtb[(((size_t)(bb * H_ + h) * DK_ + dk) * T_) + tt] =
                    (__bf16)(acc[t][i] + bvv);
            }
        } else {
            __bf16* dst = (which == 0) ? qb : kb;
#pragma unroll
            for (int i = 0; i < 8; ++i) {
                const int row = TILE_ROW(tms[t], lane, i);
                dst[(size_t)row * D_ + col] = (__bf16)(acc[t][i] + bvv);
            }
        }
    }
}

// ---------------------------------------------------------------------------
// Kernel 2: scores = (Q @ K^T) * 1/sqrt(DK), fp32 into attn region.
// TDM-staged: Q tile (128x64) and K tile (64x64) DMA'd into LDS, WMMA
// fragments read from LDS. Per (b,h): M=T, N=T, K=DK.
// grid = (T/128, T/64, B*H)
// ---------------------------------------------------------------------------
__global__ __launch_bounds__(256) void mha_scores_kernel(
    const __bf16* __restrict__ qb,
    const __bf16* __restrict__ kb,
    float* __restrict__ attn)
{
    __shared__ __align__(128) __bf16 ldsA[128 * DK_];   // 16 KB
    __shared__ __align__(128) __bf16 ldsB[64 * DK_];    //  8 KB

    const int bh = blockIdx.z;
    const int b  = bh >> 4;
    const int h  = bh & (H_ - 1);

    const int tid  = threadIdx.x;
    const int lane = tid & 31;
    const int w    = tid >> 5;
    const int ar   = lane & 15;
    const int kh   = (lane >> 4) * 16;

    // DMA the two tiles into LDS from wave 0 (TDM ignores EXEC, one op/wave)
    if (w == 0) {
        const __bf16* gA = qb + (size_t)(b * T_ + blockIdx.x * 128) * D_ + h * DK_;
        const __bf16* gB = kb + (size_t)(b * T_ + blockIdx.y * 64)  * D_ + h * DK_;
        tdm_load_2d_bf16((unsigned)(uintptr_t)ldsA, gA,
                         /*tensor_d0=*/DK_, /*tensor_d1=*/128,
                         /*stride0=*/D_, /*tile_d0=*/DK_, /*tile_d1=*/128);
        tdm_load_2d_bf16((unsigned)(uintptr_t)ldsB, gB,
                         /*tensor_d0=*/DK_, /*tensor_d1=*/64,
                         /*stride0=*/D_, /*tile_d0=*/DK_, /*tile_d1=*/64);
        __builtin_amdgcn_s_wait_tensorcnt(0);
    }
    __syncthreads();

    const int mr = (w & 3) * 32 + ar;       // local A row
    const int nr = (w >> 2) * 32 + ar;      // local B row

    v8f c00 = {}, c01 = {}, c10 = {}, c11 = {};
#pragma unroll
    for (int k0 = 0; k0 < DK_; k0 += 32) {
        const int kk = k0 + kh;
        v16bf a0 = frag_bf16(&ldsA[(mr)      * DK_ + kk]);
        v16bf a1 = frag_bf16(&ldsA[(mr + 16) * DK_ + kk]);
        v16bf b0 = frag_bf16(&ldsB[(nr)      * DK_ + kk]);
        v16bf b1 = frag_bf16(&ldsB[(nr + 16) * DK_ + kk]);
        c00 = wmma_bf16(a0, b0, c00);
        c01 = wmma_bf16(a0, b1, c01);
        c10 = wmma_bf16(a1, b0, c10);
        c11 = wmma_bf16(a1, b1, c11);
    }

    float* out = attn + (size_t)bh * T_ * T_;
    const float scale = 0.125f;             // 1/sqrt(64)
    const int m0 = blockIdx.x * 128 + (w & 3) * 32;
    const int n0 = blockIdx.y * 64  + (w >> 2) * 32;

    v8f acc[4] = { c00, c01, c10, c11 };
    int tms[4] = { m0, m0, m0 + 16, m0 + 16 };
    int tns[4] = { n0, n0 + 16, n0, n0 + 16 };
#pragma unroll
    for (int t = 0; t < 4; ++t) {
        const int col = TILE_COL(tns[t], lane);
#pragma unroll
        for (int i = 0; i < 8; ++i) {
            const int row = TILE_ROW(tms[t], lane, i);
            out[(size_t)row * T_ + col] = acc[t][i] * scale;
        }
    }
}

// ---------------------------------------------------------------------------
// Kernel 3: row-wise masked softmax over attn, in place.
// ---------------------------------------------------------------------------
__global__ __launch_bounds__(256) void mha_softmax_kernel(
    float* __restrict__ attn, const int* __restrict__ mask)
{
    __shared__ float red[256];
    const int tid = threadIdx.x;
    const size_t r = blockIdx.x;            // r = (b*H + h)*T + q
    const int qidx = (int)(r & (T_ - 1));
    const int bh   = (int)(r >> 11);
    const int b    = bh >> 4;

    float* row = attn + r * (size_t)T_;
    const int* mrow = mask + ((size_t)b * T_ + qidx) * T_;

    float s[8];
    int   m[8];
    float mx = -3.0e38f;
#pragma unroll
    for (int i = 0; i < 8; ++i) {
        const int c = tid + i * 256;
        s[i] = row[c];
        m[i] = mrow[c];
        if (!m[i]) s[i] = -3.0e38f;
        mx = fmaxf(mx, s[i]);
    }

    red[tid] = mx;
    __syncthreads();
    for (int st = 128; st > 0; st >>= 1) {
        if (tid < st) red[tid] = fmaxf(red[tid], red[tid + st]);
        __syncthreads();
    }
    mx = red[0];
    __syncthreads();

    float e[8];
    float sum = 0.f;
#pragma unroll
    for (int i = 0; i < 8; ++i) {
        e[i] = m[i] ? __expf(s[i] - mx) : 0.f;
        sum += e[i];
    }
    red[tid] = sum;
    __syncthreads();
    for (int st = 128; st > 0; st >>= 1) {
        if (tid < st) red[tid] += red[tid + st];
        __syncthreads();
    }
    sum = red[0];

    const float inv = 1.0f / sum;
#pragma unroll
    for (int i = 0; i < 8; ++i) {
        const int c = tid + i * 256;
        row[c] = e[i] * inv;
    }
}

// ---------------------------------------------------------------------------
// Kernel 4: X = attn @ V  per (b,h).  M=T, N=DK, K=T.
// attn fp32 -> bf16 via native cvt; V^T bf16 contiguous in K.
// ---------------------------------------------------------------------------
__global__ __launch_bounds__(256) void mha_av_kernel(
    const float* __restrict__ attn,
    const __bf16* __restrict__ vtb,
    __bf16* __restrict__ xb)
{
    const int bh = blockIdx.z;
    const int b  = bh >> 4;
    const int h  = bh & (H_ - 1);

    const int tid  = threadIdx.x;
    const int lane = tid & 31;
    const int w    = tid >> 5;
    const int m0   = blockIdx.x * 128 + (w & 3) * 32;
    const int n0   = (w >> 2) * 32;         // N = 64 total
    const int ar   = lane & 15;
    const int kh   = (lane >> 4) * 16;

    const float* A = attn + (size_t)bh * T_ * T_;
    const __bf16* Bm = vtb + (size_t)bh * DK_ * T_;

    const float* Am0 = A + (size_t)(m0 + ar)      * T_;
    const float* Am1 = A + (size_t)(m0 + 16 + ar) * T_;
    const __bf16* Bn0 = Bm + (size_t)(n0 + ar)      * T_;
    const __bf16* Bn1 = Bm + (size_t)(n0 + 16 + ar) * T_;

    v8f c00 = {}, c01 = {}, c10 = {}, c11 = {};
    for (int k0 = 0; k0 < T_; k0 += 32) {
        const int kk = k0 + kh;
        __builtin_prefetch(Am0 + kk + 256, 0, 0);
        v16bf a0 = frag_f32(Am0 + kk);
        v16bf a1 = frag_f32(Am1 + kk);
        v16bf b0 = frag_bf16(Bn0 + kk);
        v16bf b1 = frag_bf16(Bn1 + kk);
        c00 = wmma_bf16(a0, b0, c00);
        c01 = wmma_bf16(a0, b1, c01);
        c10 = wmma_bf16(a1, b0, c10);
        c11 = wmma_bf16(a1, b1, c11);
    }

    v8f acc[4] = { c00, c01, c10, c11 };
    int tms[4] = { m0, m0, m0 + 16, m0 + 16 };
    int tns[4] = { n0, n0 + 16, n0, n0 + 16 };
#pragma unroll
    for (int t = 0; t < 4; ++t) {
        const int col = TILE_COL(tns[t], lane);     // dk in [0,64)
#pragma unroll
        for (int i = 0; i < 8; ++i) {
            const int row = TILE_ROW(tms[t], lane, i);
            xb[((size_t)b * T_ + row) * D_ + h * DK_ + col] = (__bf16)acc[t][i];
        }
    }
}

// ---------------------------------------------------------------------------
// Kernel 5: out = X @ Wo^T + bo  (M=B*T, N=D, K=D), fp32 output.
// ---------------------------------------------------------------------------
__global__ __launch_bounds__(256) void mha_oproj_kernel(
    const __bf16* __restrict__ xb,
    const __bf16* __restrict__ wob, const float* __restrict__ bo,
    float* __restrict__ out)
{
    const int tid  = threadIdx.x;
    const int lane = tid & 31;
    const int w    = tid >> 5;
    const int m0   = blockIdx.x * 128 + (w & 3) * 32;
    const int n0   = blockIdx.y * 64  + (w >> 2) * 32;
    const int ar   = lane & 15;
    const int kh   = (lane >> 4) * 16;

    const __bf16* Am0 = xb  + (size_t)(m0 + ar)      * D_;
    const __bf16* Am1 = xb  + (size_t)(m0 + 16 + ar) * D_;
    const __bf16* Bn0 = wob + (size_t)(n0 + ar)      * D_;
    const __bf16* Bn1 = wob + (size_t)(n0 + 16 + ar) * D_;

    v8f c00 = {}, c01 = {}, c10 = {}, c11 = {};
    for (int k0 = 0; k0 < D_; k0 += 32) {
        const int kk = k0 + kh;
        __builtin_prefetch(Bn0 + kk + 256, 0, 0);
        v16bf a0 = frag_bf16(Am0 + kk);
        v16bf a1 = frag_bf16(Am1 + kk);
        v16bf b0 = frag_bf16(Bn0 + kk);
        v16bf b1 = frag_bf16(Bn1 + kk);
        c00 = wmma_bf16(a0, b0, c00);
        c01 = wmma_bf16(a0, b1, c01);
        c10 = wmma_bf16(a1, b0, c10);
        c11 = wmma_bf16(a1, b1, c11);
    }

    v8f acc[4] = { c00, c01, c10, c11 };
    int tms[4] = { m0, m0, m0 + 16, m0 + 16 };
    int tns[4] = { n0, n0 + 16, n0, n0 + 16 };
#pragma unroll
    for (int t = 0; t < 4; ++t) {
        const int col = TILE_COL(tns[t], lane);
        const float bvv = bo[col];
#pragma unroll
        for (int i = 0; i < 8; ++i) {
            const int row = TILE_ROW(tms[t], lane, i);
            out[(size_t)row * D_ + col] = acc[t][i] + bvv;
        }
    }
}

// ---------------------------------------------------------------------------
// Host-side launch
// ---------------------------------------------------------------------------
extern "C" void kernel_launch(void* const* d_in, const int* in_sizes, int n_in,
                              void* d_out, int out_size, void* d_ws, size_t ws_size,
                              hipStream_t stream) {
    (void)in_sizes; (void)n_in; (void)out_size; (void)ws_size;

    const float* query = (const float*)d_in[0];
    const float* key   = (const float*)d_in[1];
    const float* value = (const float*)d_in[2];
    const int*   mask  = (const int*)  d_in[3];
    const float* Wq = (const float*)d_in[4];
    const float* bq = (const float*)d_in[5];
    const float* Wk = (const float*)d_in[6];
    const float* bk = (const float*)d_in[7];
    const float* Wv = (const float*)d_in[8];
    const float* bv = (const float*)d_in[9];
    const float* Wo = (const float*)d_in[10];
    const float* bo = (const float*)d_in[11];

    float* out  = (float*)d_out;                         // (B,T,D)
    float* attn = out + (size_t)B_ * T_ * D_;            // (B,H,T,T)

    // Workspace layout (bf16)
    const size_t E = (size_t)B_ * T_ * D_;               // 8,388,608
    const size_t W = (size_t)D_ * D_;                    // 1,048,576
    __bf16* qx  = (__bf16*)d_ws;
    __bf16* kx  = qx  + E;
    __bf16* vx  = kx  + E;
    __bf16* wqb = vx  + E;
    __bf16* wkb = wqb + W;
    __bf16* wvb = wkb + W;
    __bf16* wob = wvb + W;
    __bf16* qb  = wob + W;
    __bf16* kb  = qb  + E;
    __bf16* vtb = kb  + E;                               // per-head transposed V
    __bf16* xb  = vtb + E;

    dim3 blk(256);

    // 0) fp32 -> bf16 pre-conversion (inputs + weights)
    mha_cvt_kernel<<<dim3((int)(E / 4 / 256)), blk, 0, stream>>>(query, qx, (int)(E / 4));
    mha_cvt_kernel<<<dim3((int)(E / 4 / 256)), blk, 0, stream>>>(key,   kx, (int)(E / 4));
    mha_cvt_kernel<<<dim3((int)(E / 4 / 256)), blk, 0, stream>>>(value, vx, (int)(E / 4));
    mha_cvt_kernel<<<dim3((int)(W / 4 / 256)), blk, 0, stream>>>(Wq, wqb, (int)(W / 4));
    mha_cvt_kernel<<<dim3((int)(W / 4 / 256)), blk, 0, stream>>>(Wk, wkb, (int)(W / 4));
    mha_cvt_kernel<<<dim3((int)(W / 4 / 256)), blk, 0, stream>>>(Wv, wvb, (int)(W / 4));
    mha_cvt_kernel<<<dim3((int)(W / 4 / 256)), blk, 0, stream>>>(Wo, wob, (int)(W / 4));

    // 1) QKV projections
    mha_qkv_kernel<<<dim3(64, 16, 3), blk, 0, stream>>>(
        qx, kx, vx, wqb, wkb, wvb, bq, bk, bv, qb, kb, vtb);

    // 2) Scores -> attn region (TDM-staged tiles)
    mha_scores_kernel<<<dim3(T_ / 128, T_ / 64, B_ * H_), blk, 0, stream>>>(
        qb, kb, attn);

    // 3) Masked softmax, in place
    mha_softmax_kernel<<<dim3(B_ * H_ * T_), blk, 0, stream>>>(attn, mask);

    // 4) X = attn @ V
    mha_av_kernel<<<dim3(T_ / 128, 1, B_ * H_), blk, 0, stream>>>(attn, vtb, xb);

    // 5) out = X @ Wo^T + bo
    mha_oproj_kernel<<<dim3(64, 16, 1), blk, 0, stream>>>(xb, Wo ? wob : wob, bo, out);
}
